// DiscriptorMatchLoss_15942918603143
// MI455X (gfx1250) — compile-verified, matrix operand order checked by reading, not run on hardware
//
#include <hip/hip_runtime.h>
#include <hip/hip_bf16.h>

// Problem constants (match reference setup_inputs)
#define BB    4
#define NN    2048
#define DD    256
#define TILE  128
#define KSTEP 32
#define RAD2  64.0f   // RADIUS^2

typedef __attribute__((ext_vector_type(16))) _Float16 v16h;
typedef __attribute__((ext_vector_type(8)))  float    v8f;

union FragU { v16h v; float4 f4[2]; };

// ---------------------------------------------------------------------------
// Kernel 1: L2-normalize each feature row into f16 scratch.
// cos(a,b) = <a/|a|, b/|b|>, so the GEMM below directly yields cosines.
// ---------------------------------------------------------------------------
__global__ __launch_bounds__(256) void dml_norm(const float* __restrict__ f,
                                                _Float16* __restrict__ nf) {
  __shared__ float sred[256];
  const int row = blockIdx.x;          // 0 .. BB*NN-1
  const int t   = threadIdx.x;         // 0 .. 255 == DD-1
  const float v = f[(size_t)row * DD + t];
  sred[t] = v * v;
  __syncthreads();
  #pragma unroll
  for (int s = 128; s > 0; s >>= 1) {
    if (t < s) sred[t] += sred[t + s];
    __syncthreads();
  }
  const float scale = rsqrtf(fmaxf(sred[0], 1e-16f));
  nf[(size_t)row * DD + t] = (_Float16)(v * scale);
}

// ---------------------------------------------------------------------------
// Kernel 2: zero global accumulators (sum, count) in workspace.
// ---------------------------------------------------------------------------
__global__ void dml_zero(float* gsum, unsigned* gcnt) {
  *gsum = 0.0f;
  *gcnt = 0u;
}

// ---------------------------------------------------------------------------
// Kernel 3: per-(i,j)-pair tiled cosine GEMM via v_wmma_f32_16x16x32_f16,
// fused with the radius mask and masked-sum reduction.
//   gridDim = (NN/TILE, NN/TILE, BB*BB), block = 256 threads (8 waves).
//   Wave w owns a 32x64 sub-tile: rows 32*(w&3), cols 64*(w>>2).
// ---------------------------------------------------------------------------
__global__ __launch_bounds__(256)
void dml_main(const _Float16* __restrict__ nf,
              const float* __restrict__ pts_src,   // [BB,NN,2]
              const float* __restrict__ pts_dst,   // [BB,BB,NN,2]
              const int* __restrict__ hp, const int* __restrict__ wp,
              float* __restrict__ gsum, unsigned* __restrict__ gcnt) {
  __shared__ __align__(16) _Float16 As[TILE * KSTEP];  // 8 KB
  __shared__ __align__(16) _Float16 Bs[TILE * KSTEP];  // 8 KB
  __shared__ float    redS;
  __shared__ unsigned redC;

  const int tid = threadIdx.x;
  if (tid == 0) { redS = 0.0f; redC = 0u; }

  const int p  = blockIdx.z;
  const int i  = p >> 2;               // dst batch (p / BB)
  const int j  = p & (BB - 1);         // src batch (p % BB)
  const int n0 = blockIdx.x * TILE;    // row offset within N
  const int m0 = blockIdx.y * TILE;    // col offset within N

  // A rows come from features[j, n0..], B "columns" from features[i, m0..].
  const _Float16* Arow = nf + (size_t)(j * NN + n0) * DD;
  const _Float16* Brow = nf + (size_t)(i * NN + m0) * DD;

  const int lane = tid & 31;
  const int wave = tid >> 5;
  const int rblk = wave & 3;           // 32-row block within tile
  const int cblk = wave >> 2;          // 64-col block within tile
  const int lrow = lane & 15;
  const int kb   = (lane < 16) ? 0 : 8;  // 16-bit WMMA lane K split

  v8f acc[2][4];
  #pragma unroll
  for (int rt = 0; rt < 2; ++rt)
    #pragma unroll
    for (int ct = 0; ct < 4; ++ct)
      acc[rt][ct] = (v8f){0.f, 0.f, 0.f, 0.f, 0.f, 0.f, 0.f, 0.f};

  for (int ks = 0; ks < DD; ks += KSTEP) {
    __syncthreads();
    // Cooperative global -> LDS staging: 128 rows x 32 halves per tile.
    #pragma unroll
    for (int it = 0; it < 2; ++it) {
      const int idx = tid + it * 256;            // 0 .. 511 (16B chunks)
      const int row = idx >> 2;
      const int ch  = (idx & 3) * 8;             // half-element offset
      *(float4*)&As[row * KSTEP + ch] =
          *(const float4*)&Arow[(size_t)row * DD + ks + ch];
      *(float4*)&Bs[row * KSTEP + ch] =
          *(const float4*)&Brow[(size_t)row * DD + ks + ch];
    }
    __syncthreads();

    // Load fragments (16-bit A-matrix layout; B is symmetric here since
    // B = (rows of nf[i])^T, so its lane fragment equals the A-style row).
    FragU a[2], b[4];
    #pragma unroll
    for (int rt = 0; rt < 2; ++rt) {
      const _Float16* rp = &As[(32 * rblk + 16 * rt + lrow) * KSTEP];
      a[rt].f4[0] = *(const float4*)(rp + kb);
      a[rt].f4[1] = *(const float4*)(rp + kb + 16);
    }
    #pragma unroll
    for (int ct = 0; ct < 4; ++ct) {
      const _Float16* rp = &Bs[(64 * cblk + 16 * ct + lrow) * KSTEP];
      b[ct].f4[0] = *(const float4*)(rp + kb);
      b[ct].f4[1] = *(const float4*)(rp + kb + 16);
    }

    #pragma unroll
    for (int rt = 0; rt < 2; ++rt)
      #pragma unroll
      for (int ct = 0; ct < 4; ++ct)
        acc[rt][ct] = __builtin_amdgcn_wmma_f32_16x16x32_f16(
            false, a[rt].v, false, b[ct].v, (short)0, acc[rt][ct],
            false, false);
  }

  // ------------------- epilogue: mask + masked reduction -------------------
  __syncthreads();   // all WMMA fragment reads done; reuse LDS for points
  float* psS = (float*)As;   // 128 x float2 denormalized src points
  float* pdS = (float*)Bs;   // 128 x float2 denormalized dst points

  const int H = *hp, W = *wp;
  const float sx = 0.5f * (float)(W - 1);
  const float sy = 0.5f * (float)(H - 1);

  if (tid < TILE) {
    const size_t base = (size_t)(i * NN + n0 + tid) * 2;
    psS[2 * tid + 0] = (pts_src[base + 0] + 1.0f) * sx;
    psS[2 * tid + 1] = (pts_src[base + 1] + 1.0f) * sy;
  } else if (tid < 2 * TILE) {
    const int t = tid - TILE;
    const size_t base = ((size_t)((i * BB + j) * NN) + (m0 + t)) * 2;
    pdS[2 * t + 0] = (pts_dst[base + 0] + 1.0f) * sx;
    pdS[2 * t + 1] = (pts_dst[base + 1] + 1.0f) * sy;
  }
  __syncthreads();

  float pmx[4], pmy[4];
  #pragma unroll
  for (int ct = 0; ct < 4; ++ct) {
    const int ml = 64 * cblk + 16 * ct + lrow;   // N = lane&15 in C layout
    pmx[ct] = pdS[2 * ml + 0];
    pmy[ct] = pdS[2 * ml + 1];
  }

  float    lsum = 0.0f;
  unsigned lcnt = 0u;
  const int mhi = (lane < 16) ? 0 : 8;           // C layout: M = v (+8 hi)
  #pragma unroll
  for (int rt = 0; rt < 2; ++rt) {
    #pragma unroll
    for (int v = 0; v < 8; ++v) {
      const int nl = 32 * rblk + 16 * rt + v + mhi;
      const float px = psS[2 * nl + 0];
      const float py = psS[2 * nl + 1];
      #pragma unroll
      for (int ct = 0; ct < 4; ++ct) {
        const float dx = px - pmx[ct];
        const float dy = py - pmy[ct];
        if (dx * dx + dy * dy <= RAD2) {
          lsum += 1.0f - acc[rt][ct][v];
          lcnt += 1u;
        }
      }
    }
  }

  atomicAdd(&redS, lsum);        // ds_add_f32
  atomicAdd(&redC, lcnt);
  __syncthreads();
  if (tid == 0) {
    atomicAdd(gsum, redS);       // global_atomic_add_f32
    atomicAdd(gcnt, redC);
  }
}

// ---------------------------------------------------------------------------
// Kernel 4: finalize scalar output = sum / max(cnt, 1)
// ---------------------------------------------------------------------------
__global__ void dml_fin(const float* gsum, const unsigned* gcnt,
                        float* __restrict__ out) {
  const float c = (float)(*gcnt);
  out[0] = (*gsum) / fmaxf(c, 1.0f);
}

// ---------------------------------------------------------------------------
extern "C" void kernel_launch(void* const* d_in, const int* in_sizes, int n_in,
                              void* d_out, int out_size, void* d_ws, size_t ws_size,
                              hipStream_t stream) {
  const float* features = (const float*)d_in[0];   // [BB,NN,DD] f32
  const float* pts_src  = (const float*)d_in[1];   // [BB,NN,2]  f32
  const float* pts_dst  = (const float*)d_in[2];   // [BB,BB,NN,2] f32
  // d_in[3] = invis_idx (unused by the reference loss)
  const int*   hp       = (const int*)d_in[4];     // height (scalar)
  const int*   wp       = (const int*)d_in[5];     // width  (scalar)

  // Workspace layout: [0] float sum accumulator, [1] uint count, then
  // (256-byte aligned) normalized f16 features [BB*NN, DD] = 4 MB.
  float*     gsum = (float*)d_ws;
  unsigned*  gcnt = (unsigned*)d_ws + 1;
  _Float16*  nf   = (_Float16*)((char*)d_ws + 256);

  dml_norm<<<BB * NN, 256, 0, stream>>>(features, nf);
  dml_zero<<<1, 1, 0, stream>>>(gsum, gcnt);

  dim3 grid(NN / TILE, NN / TILE, BB * BB);        // 16 x 16 x 16 blocks
  dml_main<<<grid, 256, 0, stream>>>(nf, pts_src, pts_dst, hp, wp, gsum, gcnt);

  dml_fin<<<1, 1, 0, stream>>>(gsum, gcnt, (float*)d_out);
}
